// MultiHeadAttention_16209206575629
// MI455X (gfx1250) — compile-verified, matrix operand order checked by reading, not run on hardware
//
#include <hip/hip_runtime.h>

// B=2, S=2048, E=1024, H=16, D=64
typedef __attribute__((ext_vector_type(16))) __bf16 bf16x16;
typedef __attribute__((ext_vector_type(8)))  float  f32x8;
typedef __attribute__((ext_vector_type(4)))  unsigned int u32x4;
typedef __attribute__((ext_vector_type(8)))  int i32x8;
typedef __attribute__((ext_vector_type(4)))  int i32x4;

#if defined(__has_builtin)
#  if __has_builtin(__builtin_amdgcn_tensor_load_to_lds)
#    define HAVE_TDM 1
#  endif
#endif
#ifndef HAVE_TDM
#  define HAVE_TDM 0
#endif

__device__ __forceinline__ unsigned short f2bf(float f) {
  unsigned int u = __builtin_bit_cast(unsigned int, f);
  u += 0x7FFFu + ((u >> 16) & 1u);   // round-to-nearest-even
  return (unsigned short)(u >> 16);
}

__device__ __forceinline__ f32x8 wmma_bf16(bf16x16 a, bf16x16 b, f32x8 c) {
  return __builtin_amdgcn_wmma_f32_16x16x32_bf16(false, a, false, b, (short)0, c, false, false);
}

// A/B fragment loader for 16x16x32 bf16 WMMA from an LDS tile stored
// "16 outer rows x 32 K-contiguous" with row stride `stride` (ushorts).
// lanes 0-15 -> row=lane, K={0..7,16..23}; lanes 16-31 -> row=lane-16, K={8..15,24..31}.
__device__ __forceinline__ bf16x16 ld_frag(const unsigned short* p, int stride) {
  const int lane = threadIdx.x & 31;
  const int m = lane & 15, hf = lane >> 4;
  const unsigned short* r0 = p + m * stride + hf * 8;
  union { bf16x16 v; u32x4 q[2]; } f;
  f.q[0] = *(const u32x4*)(r0);
  f.q[1] = *(const u32x4*)(r0 + 16);
  return f.v;
}

__device__ __forceinline__ float rmax16(float v) {
#pragma unroll
  for (int o = 1; o < 16; o <<= 1) v = fmaxf(v, __shfl_xor(v, o, 32));
  return v;
}
__device__ __forceinline__ float rsum16(float v) {
#pragma unroll
  for (int o = 1; o < 16; o <<= 1) v += __shfl_xor(v, o, 32);
  return v;
}

#if HAVE_TDM
// TDM 2D tile load: D# group0/group1 per CDNA5 ISA 8.3/8.4.
// data_size = 2 bytes (bf16). Padding reproduces the LDS row padding the
// fragment loader expects (pad_amount DWORDs inserted every pad_interval).
__device__ __forceinline__ void tdm_load_2d(
    unsigned int lds_byte_off, const void* gsrc,
    unsigned int tensor_d0, unsigned int tensor_d1,
    unsigned int tile_d0, unsigned int tile_d1,
    unsigned int stride_d0,
    unsigned int pad_interval_enc, unsigned int pad_amount_enc) {
  unsigned long long ga = (unsigned long long)gsrc;
  u32x4 g0;
  g0[0] = 1u;                                        // count=1, user descriptor
  g0[1] = lds_byte_off;                              // lds_addr
  g0[2] = (unsigned int)(ga & 0xFFFFFFFFu);          // global_addr[31:0]
  g0[3] = (unsigned int)((ga >> 32) & 0x01FFFFFFu)   // global_addr[56:32]
        | (2u << 30);                                // type = 2 (image)
  i32x8 g1;
  g1[0] = (int)((1u << 16)                           // data_size = 2 bytes
        | (1u << 20)                                 // pad_enable
        | (pad_interval_enc << 22) | (pad_amount_enc << 25));
  g1[1] = (int)((tensor_d0 & 0xFFFFu) << 16);        // tensor_dim0[15:0]
  g1[2] = (int)(((tensor_d0 >> 16) & 0xFFFFu) | ((tensor_d1 & 0xFFFFu) << 16));
  g1[3] = (int)(((tensor_d1 >> 16) & 0xFFFFu) | ((tile_d0 & 0xFFFFu) << 16));
  g1[4] = (int)(tile_d1 & 0xFFFFu);                  // tile_dim1; tile_dim2=0
  g1[5] = (int)stride_d0;                            // tensor_dim0_stride[31:0]
  g1[6] = 0;
  g1[7] = 0;
  i32x4 z4 = {0, 0, 0, 0};
#if __clang_major__ >= 23
  i32x8 z8 = {0, 0, 0, 0, 0, 0, 0, 0};
  __builtin_amdgcn_tensor_load_to_lds(g0, g1, z4, z4, z8, 0);
#else
  __builtin_amdgcn_tensor_load_to_lds(g0, g1, z4, z4, 0);
#endif
}
__device__ __forceinline__ unsigned int lds_off(const void* p) {
  return (unsigned int)(unsigned long long)p;        // generic ptr low 32 = LDS offset
}
#endif

__global__ void cvt_bf16(const float* __restrict__ in, unsigned short* __restrict__ out, int n) {
  int i = (blockIdx.x * blockDim.x + threadIdx.x) * 4;
  if (i + 3 < n) {
    float4 v = *(const float4*)(in + i);
    out[i + 0] = f2bf(v.x); out[i + 1] = f2bf(v.y);
    out[i + 2] = f2bf(v.z); out[i + 3] = f2bf(v.w);
  }
}

// 128x128 block tile, 8 waves (wave32) each computing 32(M)x64(N). K = 1024.
// Double-buffered LDS; TDM prefetches tile t+1 while tile t is consumed.
// EPI 0: QKV scatter epilogue (bf16 planes). EPI 1: +bias fp32 to f_out.
template <int EPI>
__global__ __launch_bounds__(256) void gemm128(
    const unsigned short* __restrict__ A, const unsigned short* __restrict__ Bm,
    const float* __restrict__ bias, unsigned short* __restrict__ qkv_out,
    float* __restrict__ f_out, int nrows) {
  __shared__ unsigned short As[2][128][40];
  __shared__ unsigned short Bs[2][128][40];
  const int m0 = blockIdx.y * 128, n0 = blockIdx.x * 128;
  const int wv = threadIdx.x >> 5, wm = wv >> 1, wn = wv & 1;
  const int lane = threadIdx.x & 31, col = lane & 15, rb = (lane >> 4) * 8;
  f32x8 acc[2][4] = {};
#if HAVE_TDM
  if (wv == 0) {  // prologue: DMA tile 0
    // 128 rows x 32 bf16; row = 16 DWORDs (enc 3), pad 4 DWORDs (enc 3) -> [128][40]
    tdm_load_2d(lds_off(&As[0][0][0]), A  + (size_t)m0 * 1024, 1024, 4096, 32, 128, 1024, 3, 3);
    tdm_load_2d(lds_off(&Bs[0][0][0]), Bm + (size_t)n0 * 1024, 1024, (unsigned)nrows, 32, 128, 1024, 3, 3);
  }
#else
  const int tr = threadIdx.x >> 1, tc = (threadIdx.x & 1) * 16;
  const unsigned short* gA = A  + (size_t)(m0 + tr) * 1024 + tc;
  const unsigned short* gB = Bm + (size_t)(n0 + tr) * 1024 + tc;
  u32x4 a0 = *(const u32x4*)(gA);
  u32x4 a1 = *(const u32x4*)(gA + 8);
  u32x4 b0 = *(const u32x4*)(gB);
  u32x4 b1 = *(const u32x4*)(gB + 8);
#endif
  for (int t = 0; t < 32; t++) {
    const int cur = t & 1, nxt = cur ^ 1;
    __syncthreads();  // readers of buffer[nxt] (iteration t-1) are done
#if HAVE_TDM
    if (wv == 0) {
      if (t + 1 < 32) {
        const int kk = (t + 1) * 32;
        tdm_load_2d(lds_off(&As[nxt][0][0]), A  + (size_t)m0 * 1024 + kk,
                    1024, 4096, 32, 128, 1024, 3, 3);
        tdm_load_2d(lds_off(&Bs[nxt][0][0]), Bm + (size_t)n0 * 1024 + kk,
                    1024, (unsigned)nrows, 32, 128, 1024, 3, 3);
        __builtin_amdgcn_s_wait_tensorcnt(2);  // tile t complete, t+1 in flight
      } else {
        __builtin_amdgcn_s_wait_tensorcnt(0);
      }
    }
#else
    *(u32x4*)&As[cur][tr][tc] = a0; *(u32x4*)&As[cur][tr][tc + 8] = a1;
    *(u32x4*)&Bs[cur][tr][tc] = b0; *(u32x4*)&Bs[cur][tr][tc + 8] = b1;
    if (t + 1 < 32) {  // issue next tile's global loads under this tile's compute
      const int kk = (t + 1) * 32;
      a0 = *(const u32x4*)(gA + kk);
      a1 = *(const u32x4*)(gA + kk + 8);
      b0 = *(const u32x4*)(gB + kk);
      b1 = *(const u32x4*)(gB + kk + 8);
    }
#endif
    __syncthreads();  // buffer[cur] published
    bf16x16 af[2], bfm[4];
#pragma unroll
    for (int i = 0; i < 2; i++) af[i] = ld_frag(&As[cur][wm * 32 + i * 16][0], 40);
#pragma unroll
    for (int j = 0; j < 4; j++) bfm[j] = ld_frag(&Bs[cur][wn * 64 + j * 16][0], 40);
#pragma unroll
    for (int i = 0; i < 2; i++)
#pragma unroll
      for (int j = 0; j < 4; j++)
        acc[i][j] = wmma_bf16(af[i], bfm[j], acc[i][j]);
  }
#pragma unroll
  for (int i = 0; i < 2; i++)
#pragma unroll
    for (int j = 0; j < 4; j++) {
      const int n = n0 + wn * 64 + j * 16 + col;
      const float bv = bias[n];
#pragma unroll
      for (int r = 0; r < 8; r++) {
        const int g = m0 + wm * 32 + i * 16 + rb + r;
        const float v = acc[i][j][r] + bv;
        if (EPI == 0) {
          const int t2 = n >> 10, h = (n >> 6) & 15, d = n & 63;
          const int b = g >> 11, s = g & 2047;
          qkv_out[(((size_t)(t2 * 2 + b) * 16 + h) * 2048 + s) * 64 + d] = f2bf(v);
        } else {
          f_out[(size_t)g * 1024 + n] = v;
        }
      }
    }
}

// Flash attention with multiplicative exp(-|a|*|i-j|) decay on logits.
// grid = (B*H, S/64), 128 threads = 4 waves; wave w owns query rows [q0+16w, q0+16w+16).
// K tiles: TDM double-buffer prefetch. V tiles: register-pipelined transpose copy.
__global__ __launch_bounds__(128) void attn_kernel(
    const unsigned short* __restrict__ qkv, const float* __restrict__ dd,
    unsigned short* __restrict__ obf) {
  __shared__ unsigned short Qs[64][72];
  __shared__ unsigned short Ks[2][64][72];
  __shared__ unsigned short Vt[2][64][72];  // transposed: Vt[buf][d][j]
  __shared__ unsigned short Ps[64][72];     // 4 waves x 16 rows of probabilities
  const int bh = blockIdx.x;
  const int b = bh >> 4, h = bh & 15;
  const int q0 = blockIdx.y * 64;
  const int wv = threadIdx.x >> 5;
  const int lane = threadIdx.x & 31, col = lane & 15, rb = (lane >> 4) * 8;
  const float absa = fabsf(dd[0]);

  const size_t head = ((size_t)b * 16 + h) * (size_t)2048 * 64;
  const unsigned short* Qg = qkv + head + (size_t)q0 * 64;
  const unsigned short* Kg = qkv + (size_t)4194304 + head;  // t=1 plane
  const unsigned short* Vg = qkv + (size_t)8388608 + head;  // t=2 plane

  const int vr = threadIdx.x >> 1, vcb = (threadIdx.x & 1) * 32;
  union V32 { u32x4 q[4]; unsigned short u[32]; };
  V32 vreg;
#pragma unroll
  for (int e = 0; e < 4; e++)
    vreg.q[e] = ((const u32x4*)(Vg + (size_t)vr * 64 + vcb))[e];

#if HAVE_TDM
  if (wv == 0) {
    // 64 rows x 64 bf16; row = 32 DWORDs (enc 4), pad 4 DWORDs (enc 3) -> [64][72]
    tdm_load_2d(lds_off(&Qs[0][0]),    Qg, 64, 2048, 64, 64, 64, 4, 3);
    tdm_load_2d(lds_off(&Ks[0][0][0]), Kg, 64, 2048, 64, 64, 64, 4, 3);
  }
#else
  {
    const u32x4* src = (const u32x4*)(Qg + (size_t)vr * 64 + vcb);
#pragma unroll
    for (int e = 0; e < 4; e++) *(u32x4*)&Qs[vr][vcb + e * 8] = src[e];
  }
  V32 kreg;
#pragma unroll
  for (int e = 0; e < 4; e++)
    kreg.q[e] = ((const u32x4*)(Kg + (size_t)vr * 64 + vcb))[e];
#endif

  float mrun[8], lrun[8];
#pragma unroll
  for (int r = 0; r < 8; r++) { mrun[r] = -3.0e38f; lrun[r] = 0.f; }
  f32x8 oacc[4] = {};
  bf16x16 qa[2];
  bool qa_ready = false;

  for (int jb = 0; jb < 32; jb++) {
    const int cur = jb & 1, nxt = cur ^ 1;
    const int j0 = jb * 64;
    __syncthreads();  // readers of buffer[nxt] (iteration jb-1) done
#if HAVE_TDM
    if (wv == 0 && jb + 1 < 32)
      tdm_load_2d(lds_off(&Ks[nxt][0][0]), Kg + (size_t)(j0 + 64) * 64,
                  64, 2048, 64, 64, 64, 4, 3);
#endif
    V32 vnext{};
    if (jb + 1 < 32) {  // issue next V block's loads under this block's compute
#pragma unroll
      for (int e = 0; e < 4; e++)
        vnext.q[e] = ((const u32x4*)(Vg + (size_t)(j0 + 64 + vr) * 64 + vcb))[e];
    }
#pragma unroll
    for (int e = 0; e < 32; e++) Vt[cur][vcb + e][vr] = vreg.u[e];  // transpose
#if HAVE_TDM
    if (wv == 0) {
      if (jb + 1 < 32) __builtin_amdgcn_s_wait_tensorcnt(1);  // K(jb) done
      else             __builtin_amdgcn_s_wait_tensorcnt(0);
    }
#else
#pragma unroll
    for (int e = 0; e < 4; e++) *(u32x4*)&Ks[cur][vr][vcb + e * 8] = kreg.q[e];
    if (jb + 1 < 32) {
#pragma unroll
      for (int e = 0; e < 4; e++)
        kreg.q[e] = ((const u32x4*)(Kg + (size_t)(j0 + 64 + vr) * 64 + vcb))[e];
    }
#endif
    __syncthreads();  // buffers[cur] published (and Qs on first pass)

    if (!qa_ready) {  // Q fragments once (Qs static after first barrier)
      qa[0] = ld_frag(&Qs[wv * 16][0], 72);
      qa[1] = ld_frag(&Qs[wv * 16][32], 72);
      qa_ready = true;
    }
    // S = Q K^T  (16 q-rows x 64 keys per wave)
    f32x8 sacc[4] = {};
#pragma unroll
    for (int ct = 0; ct < 4; ct++) {
      bf16x16 kb0 = ld_frag(&Ks[cur][ct * 16][0], 72);
      bf16x16 kb1 = ld_frag(&Ks[cur][ct * 16][32], 72);
      sacc[ct] = wmma_bf16(qa[0], kb0, sacc[ct]);
      sacc[ct] = wmma_bf16(qa[1], kb1, sacc[ct]);
    }
    // logits with multiplicative decay: z = s * 0.125 * exp(-|a|*|i-j|)
#pragma unroll
    for (int ct = 0; ct < 4; ct++)
#pragma unroll
      for (int r = 0; r < 8; r++) {
        const int qi = q0 + wv * 16 + rb + r;
        const int kj = j0 + ct * 16 + col;
        sacc[ct][r] = sacc[ct][r] * 0.125f * __expf(-absa * fabsf((float)(qi - kj)));
      }
    float mnew[8], al[8], rs[8];
#pragma unroll
    for (int r = 0; r < 8; r++) {
      float v = fmaxf(fmaxf(sacc[0][r], sacc[1][r]), fmaxf(sacc[2][r], sacc[3][r]));
      mnew[r] = fmaxf(mrun[r], rmax16(v));
      al[r] = __expf(mrun[r] - mnew[r]);
      rs[r] = 0.f;
    }
#pragma unroll
    for (int ct = 0; ct < 4; ct++)
#pragma unroll
      for (int r = 0; r < 8; r++) {
        const float p = __expf(sacc[ct][r] - mnew[r]);
        rs[r] += p;
        Ps[wv * 16 + rb + r][ct * 16 + col] = f2bf(p);
      }
#pragma unroll
    for (int r = 0; r < 8; r++) {
      lrun[r] = al[r] * lrun[r] + rsum16(rs[r]);
      mrun[r] = mnew[r];
    }
#pragma unroll
    for (int ct = 0; ct < 4; ct++)
#pragma unroll
      for (int r = 0; r < 8; r++) oacc[ct][r] *= al[r];
    __syncthreads();  // P visible for fragment loads
    // O += P V
#pragma unroll
    for (int c = 0; c < 2; c++) {
      bf16x16 pa = ld_frag(&Ps[wv * 16][c * 32], 72);
#pragma unroll
      for (int ct = 0; ct < 4; ct++) {
        bf16x16 vb = ld_frag(&Vt[cur][ct * 16][c * 32], 72);
        oacc[ct] = wmma_bf16(pa, vb, oacc[ct]);
      }
    }
    vreg = vnext;
  }
#pragma unroll
  for (int r = 0; r < 8; r++) lrun[r] = 1.0f / lrun[r];
#pragma unroll
  for (int ct = 0; ct < 4; ct++)
#pragma unroll
    for (int r = 0; r < 8; r++) {
      const int s = q0 + wv * 16 + rb + r;
      const size_t off = ((size_t)b * 2048 + s) * 1024 + h * 64 + ct * 16 + col;
      obf[off] = f2bf(oacc[ct][r] * lrun[r]);
    }
}

extern "C" void kernel_launch(void* const* d_in, const int* in_sizes, int n_in,
                              void* d_out, int out_size, void* d_ws, size_t ws_size,
                              hipStream_t stream) {
  (void)in_sizes; (void)n_in; (void)out_size; (void)ws_size;
  const float* x      = (const float*)d_in[0];
  const float* Wqkv_w = (const float*)d_in[1];
  const float* Wqkv_b = (const float*)d_in[2];
  const float* out_w  = (const float*)d_in[3];
  const float* out_b  = (const float*)d_in[4];
  const float* dd     = (const float*)d_in[5];

  char* ws = (char*)d_ws;                                   // bytes
  unsigned short* xbf   = (unsigned short*)(ws);            //  8,388,608 B
  unsigned short* wqbf  = (unsigned short*)(ws + 8388608);  //  6,291,456 B
  unsigned short* owbf  = (unsigned short*)(ws + 14680064); //  2,097,152 B
  unsigned short* qkvws = (unsigned short*)(ws + 16777216); // 25,165,824 B
  unsigned short* obf   = (unsigned short*)(ws + 41943040); //  8,388,608 B

  cvt_bf16<<<4096, 256, 0, stream>>>(x,      xbf,  4194304);
  cvt_bf16<<<3072, 256, 0, stream>>>(Wqkv_w, wqbf, 3145728);
  cvt_bf16<<<1024, 256, 0, stream>>>(out_w,  owbf, 1048576);

  // QKV: (4096 x 3072) = xbf (4096x1024) * Wqkv^T, scatter into Q/K/V planes
  gemm128<0><<<dim3(24, 32), 256, 0, stream>>>(xbf, wqbf, Wqkv_b, qkvws, nullptr, 3072);
  // Flash attention per (b,h) and 64-row query block
  attn_kernel<<<dim3(32, 32), 128, 0, stream>>>(qkvws, dd, obf);
  // Output projection: (4096 x 1024) fp32 + bias
  gemm128<1><<<dim3(8, 32), 256, 0, stream>>>(obf, owbf, out_b, nullptr, (float*)d_out, 1024);
}
